// MapEncoder_34883724378167
// MI455X (gfx1250) — matrix operand order repaired and checked by reading, not compiled, not run on hardware
//
#include <hip/hip_runtime.h>
#include <hip/hip_bf16.h>

// ---------------------------------------------------------------------------
// MapEncoder for MI455X (gfx1250): bf16 WMMA GEMMs + banded flash attention,
// with CDNA5 async global->LDS staging (GLOBAL_LOAD_ASYNC_TO_LDS_B128).
// B=8 T=1024 IN=64 E=512 H=8 DH=64 FF=2048 L=4, window = +/-128 (hardcoded).
// ---------------------------------------------------------------------------

typedef __attribute__((ext_vector_type(16))) __bf16 v16bf;
typedef __attribute__((ext_vector_type(8)))  float  v8f;

static constexpr int Bn  = 8;
static constexpr int Tn  = 1024;
static constexpr int INn = 64;
static constexpr int En  = 512;
static constexpr int Hn  = 8;
static constexpr int DHn = 64;
static constexpr int FFn = 2048;
static constexpr int Ln  = 4;
static constexpr int Mn  = Bn * Tn;      // 8192 rows of activation
static constexpr int WIN = 128;          // past_frames = future_frames = 128
static constexpr float SCALE = 0.125f;   // 1/sqrt(64)

union FragU {
    uint4 q[2];
    v16bf v;
    unsigned short u[16];
};

static __device__ __forceinline__ unsigned short f2bf(float f) {
    unsigned int u = __float_as_uint(f);
    unsigned int r = u + 0x7FFFu + ((u >> 16) & 1u);   // round-to-nearest-even
    return (unsigned short)(r >> 16);
}

static __device__ __forceinline__ v8f vzero8() {
    v8f z = {0.f, 0.f, 0.f, 0.f, 0.f, 0.f, 0.f, 0.f};
    return z;
}

static __device__ __forceinline__ v8f wmma_bf16(v16bf a, v16bf b, v8f c) {
    return __builtin_amdgcn_wmma_f32_16x16x32_bf16(false, a, false, b,
                                                   (short)0, c, false, false);
}

// --- CDNA5 async global->LDS copy (ASYNCcnt tracked), see CDNA5 ISA
static __device__ __forceinline__ unsigned lds_off(const void* p) {
    // generic LDS pointers carry the aperture in [63:32]; [31:0] is the
    // workgroup-relative LDS byte address the async instruction wants.
    return (unsigned)(unsigned long long)p;
}
static __device__ __forceinline__ void async_ld_b128(unsigned ldsaddr,
                                                     const void* gaddr) {
    asm volatile("global_load_async_to_lds_b128 %0, %1, off"
                 :: "v"(ldsaddr), "v"(gaddr)
                 : "memory");
}
static __device__ __forceinline__ void wait_async0() {
    asm volatile("s_wait_asynccnt 0x0" ::: "memory");
}
static __device__ __forceinline__ void wait_async4() {
    asm volatile("s_wait_asynccnt 0x4" ::: "memory");
}

// ---------------------------------------------------------------------------
// fp32 -> bf16 conversion (weights)
// ---------------------------------------------------------------------------
__global__ void cvt_bf16_kernel(const float* __restrict__ s,
                                unsigned short* __restrict__ d, int n) {
    int i = blockIdx.x * blockDim.x + threadIdx.x;
    if (i < n) d[i] = f2bf(s[i]);
}

// ---------------------------------------------------------------------------
// Input projection: x = feat @ Wp^T + bp + pos_enc   -> xf (f32) and xb (bf16)
// ---------------------------------------------------------------------------
__global__ void proj_kernel(const float* __restrict__ feat,
                            const float* __restrict__ Wp,
                            const float* __restrict__ bp,
                            const float* __restrict__ pe,
                            float* __restrict__ xf,
                            unsigned short* __restrict__ xb) {
    int bt  = blockIdx.x;
    int tid = threadIdx.x;
    int t   = bt & (Tn - 1);
    __shared__ float fr[INn];
    if (tid < INn) fr[tid] = feat[(size_t)bt * INn + tid];
    __syncthreads();
    for (int nn = tid; nn < En; nn += 256) {
        const float* w = Wp + (size_t)nn * INn;
        float acc = bp[nn] + pe[(size_t)t * En + nn];
        #pragma unroll 8
        for (int k = 0; k < INn; k++) acc += fr[k] * w[k];
        xf[(size_t)bt * En + nn] = acc;
        xb[(size_t)bt * En + nn] = f2bf(acc);
    }
}

// ---------------------------------------------------------------------------
// Tiled bf16 WMMA GEMM:  C[M,N] = act( A[M,K] @ W[N,K]^T + bias (+res) )
// Block tile 128x128, BK=32, 256 threads = 8 waves (4 along M x 2 along N).
// Double-buffered LDS filled by async global->LDS loads (no VGPR bounce).
// ---------------------------------------------------------------------------
template <bool RELU, bool RES, bool OUTF, bool OUTB>
__global__ __launch_bounds__(256)
void gemm_bf16_kernel(const unsigned short* __restrict__ A,
                      const unsigned short* __restrict__ W,
                      const float* __restrict__ bias,
                      const float* __restrict__ res,
                      float* __restrict__ outf,
                      unsigned short* __restrict__ outb,
                      int M, int N, int K) {
    __shared__ unsigned short As[2][128 * 32];
    __shared__ unsigned short Bs[2][128 * 32];

    const int m0 = blockIdx.y * 128;
    const int n0 = blockIdx.x * 128;
    const int tid  = threadIdx.x;
    const int lane = tid & 31;
    const int wid  = tid >> 5;
    const int wm   = wid & 3;     // 0..3 -> M offset wm*32
    const int wn   = wid >> 2;    // 0..1 -> N offset wn*64
    const int lh   = lane & 15;
    const int hi   = lane >> 4;

    // stage one 128x32 A tile + 128x32 W tile into LDS buffer `buf`:
    // 512 16-byte chunks per tile, 2 chunks of each per thread -> 4 asyncs.
    auto stage = [&](int k0, int buf) {
        #pragma unroll
        for (int i = 0; i < 2; i++) {
            int c   = tid + i * 256;
            int row = c >> 2;
            int off = (c & 3) * 8;
            async_ld_b128(lds_off(&As[buf][row * 32 + off]),
                          &A[(size_t)(m0 + row) * K + k0 + off]);
            async_ld_b128(lds_off(&Bs[buf][row * 32 + off]),
                          &W[(size_t)(n0 + row) * K + k0 + off]);
        }
    };

    v8f acc[2][4];
    #pragma unroll
    for (int i = 0; i < 2; i++)
        #pragma unroll
        for (int j = 0; j < 4; j++) acc[i][j] = vzero8();

    const int ntiles = K >> 5;
    stage(0, 0);
    for (int it = 0; it < ntiles; ++it) {
        const int cur = it & 1;
        if (it + 1 < ntiles) {
            stage((it + 1) << 5, cur ^ 1);   // prefetch next tile (async)
            wait_async4();                   // current tile landed
        } else {
            wait_async0();
        }
        __syncthreads();

        FragU af[2], bf[4];
        #pragma unroll
        for (int mt = 0; mt < 2; mt++) {
            int arow = wm * 32 + mt * 16 + lh;
            af[mt].q[0] = *(uint4*)&As[cur][arow * 32 + hi * 16];
            af[mt].q[1] = *(uint4*)&As[cur][arow * 32 + hi * 16 + 8];
        }
        #pragma unroll
        for (int nt = 0; nt < 4; nt++) {
            int brow = wn * 64 + nt * 16 + lh;
            bf[nt].q[0] = *(uint4*)&Bs[cur][brow * 32 + hi * 16];
            bf[nt].q[1] = *(uint4*)&Bs[cur][brow * 32 + hi * 16 + 8];
        }
        #pragma unroll
        for (int mt = 0; mt < 2; mt++)
            #pragma unroll
            for (int nt = 0; nt < 4; nt++)
                acc[mt][nt] = wmma_bf16(af[mt].v, bf[nt].v, acc[mt][nt]);

        __syncthreads();   // all reads of buffer `cur` done before reuse
    }

    // epilogue: C layout lane(<16):(M=r,N=lh)  lane(>=16):(M=8+r,N=lh)
    #pragma unroll
    for (int nt = 0; nt < 4; nt++) {
        int col  = n0 + wn * 64 + nt * 16 + lh;
        float bs = bias[col];
        #pragma unroll
        for (int mt = 0; mt < 2; mt++) {
            #pragma unroll
            for (int r = 0; r < 8; r++) {
                int row = m0 + wm * 32 + mt * 16 + hi * 8 + r;
                float v = acc[mt][nt][r] + bs;
                if (RES)  v += res[(size_t)row * N + col];
                if (RELU) v = fmaxf(v, 0.f);
                if (OUTF) outf[(size_t)row * N + col] = v;
                if (OUTB) outb[(size_t)row * N + col] = f2bf(v);
            }
        }
    }
}

// ---------------------------------------------------------------------------
// Banded flash attention, one wave (32 threads) per (b, h, 16-query tile).
// V chunks prefetched to LDS asynchronously while QK^T + softmax run.
// qkv: bf16 [B,T,3E]  ->  attn: bf16 [B,T,E]
// ---------------------------------------------------------------------------
__global__ __launch_bounds__(32)
void attn_kernel(const unsigned short* __restrict__ qkv,
                 unsigned short* __restrict__ attn) {
    const int b  = blockIdx.z;
    const int h  = blockIdx.y;
    const int q0 = blockIdx.x * 16;
    const int lane = threadIdx.x;
    const int lh   = lane & 15;
    const int hi   = lane >> 4;

    __shared__ unsigned short Vs[32 * 64];  // V chunk: 32 keys x 64 dims
    __shared__ unsigned short Ps[16 * 32];  // P tile reshaping buffer

    // Q A-fragments (d 0..31 and 32..63)
    FragU qf[2];
    {
        size_t qbase = ((size_t)(b * Tn + q0 + lh)) * (3 * En) + h * DHn;
        #pragma unroll
        for (int dc = 0; dc < 2; dc++) {
            qf[dc].q[0] = *(const uint4*)&qkv[qbase + dc * 32 + hi * 16];
            qf[dc].q[1] = *(const uint4*)&qkv[qbase + dc * 32 + hi * 16 + 8];
        }
    }

    float mrow[8], lrow[8];
    v8f O[4];
    #pragma unroll
    for (int r = 0; r < 8; r++) { mrow[r] = -1e30f; lrow[r] = 0.f; }
    #pragma unroll
    for (int nt = 0; nt < 4; nt++) O[nt] = vzero8();

    const int jlo = (q0 - WIN > 0 ? q0 - WIN : 0) & ~31;
    const int jhi = (q0 + 15 + WIN < Tn - 1 ? q0 + 15 + WIN : Tn - 1);

    for (int jc = jlo; jc <= jhi; jc += 32) {
        // ---- kick off async V-chunk staging (lands while we do QK^T) ----
        {
            int jv  = jc + lane;
            int jvc = jv < Tn ? jv : Tn - 1;
            const unsigned short* vsrc =
                &qkv[((size_t)(b * Tn + jvc)) * (3 * En) + 2 * En + h * DHn];
            unsigned vb = lds_off(&Vs[lane * 64]);
            #pragma unroll
            for (int i = 0; i < 8; i++)
                async_ld_b128(vb + i * 16, vsrc + i * 8);
        }

        // ---- S = Q @ K^T for 32-key chunk (two 16-col C tiles) ----
        v8f s[2];
        s[0] = vzero8();
        s[1] = vzero8();
        #pragma unroll
        for (int jn = 0; jn < 2; jn++) {
            int j   = jc + jn * 16 + lh;
            int jcl = j < Tn ? j : Tn - 1;
            size_t kb = ((size_t)(b * Tn + jcl)) * (3 * En) + En + h * DHn;
            #pragma unroll
            for (int dc = 0; dc < 2; dc++) {
                FragU kf;
                kf.q[0] = *(const uint4*)&qkv[kb + dc * 32 + hi * 16];
                kf.q[1] = *(const uint4*)&qkv[kb + dc * 32 + hi * 16 + 8];
                s[jn] = wmma_bf16(qf[dc].v, kf.v, s[jn]);
            }
        }

        // ---- mask + scale + online softmax (rows live across 16 lanes) ----
        #pragma unroll
        for (int r = 0; r < 8; r++) {
            int i  = q0 + hi * 8 + r;
            int j0 = jc + lh, j1 = jc + 16 + lh;
            bool ok0 = (j0 >= i - WIN) && (j0 <= i + WIN) && (j0 < Tn);
            bool ok1 = (j1 >= i - WIN) && (j1 <= i + WIN) && (j1 < Tn);
            float v0 = ok0 ? s[0][r] * SCALE : -1e30f;
            float v1 = ok1 ? s[1][r] * SCALE : -1e30f;

            float mx = fmaxf(v0, v1);
            #pragma unroll
            for (int o = 1; o < 16; o <<= 1) mx = fmaxf(mx, __shfl_xor(mx, o, 32));
            float mn   = fmaxf(mrow[r], mx);
            float corr = __expf(mrow[r] - mn);
            mrow[r] = mn;

            float p0 = ok0 ? __expf(v0 - mn) : 0.f;
            float p1 = ok1 ? __expf(v1 - mn) : 0.f;
            float ps = p0 + p1;
            #pragma unroll
            for (int o = 1; o < 16; o <<= 1) ps += __shfl_xor(ps, o, 32);
            lrow[r] = lrow[r] * corr + ps;

            #pragma unroll
            for (int nt = 0; nt < 4; nt++) O[nt][r] *= corr;

            Ps[(hi * 8 + r) * 32 + lh]      = f2bf(p0);
            Ps[(hi * 8 + r) * 32 + 16 + lh] = f2bf(p1);
        }

        // ---- P (C layout in LDS) -> A-fragment ----
        FragU pf;
        pf.q[0] = *(uint4*)&Ps[lh * 32 + hi * 16];
        pf.q[1] = *(uint4*)&Ps[lh * 32 + hi * 16 + 8];

        // ---- V staging must have landed; gather B-fragments from LDS ----
        wait_async0();
        #pragma unroll
        for (int nt = 0; nt < 4; nt++) {
            FragU vf;
            #pragma unroll
            for (int kk = 0; kk < 16; kk++)
                vf.u[kk] = Vs[(hi * 16 + kk) * 64 + nt * 16 + lh];
            O[nt] = wmma_bf16(pf.v, vf.v, O[nt]);
        }
    }

    // ---- normalize and store bf16 ----
    #pragma unroll
    for (int nt = 0; nt < 4; nt++) {
        int col = h * DHn + nt * 16 + lh;
        #pragma unroll
        for (int r = 0; r < 8; r++) {
            int t = q0 + hi * 8 + r;
            float inv = lrow[r] > 0.f ? 1.f / lrow[r] : 0.f;
            attn[((size_t)(b * Tn + t)) * En + col] = f2bf(O[nt][r] * inv);
        }
    }
}

// ---------------------------------------------------------------------------
// LayerNorm over E=512; one block (256 threads) per row; dual f32+bf16 out.
// ---------------------------------------------------------------------------
__global__ __launch_bounds__(256)
void ln_kernel(const float* __restrict__ in,
               const float* __restrict__ g,
               const float* __restrict__ bb,
               float* __restrict__ outf,
               unsigned short* __restrict__ outb) {
    int row = blockIdx.x;
    int tid = threadIdx.x;
    const float* p = in + (size_t)row * En;
    float v0 = p[tid], v1 = p[tid + 256];
    float s = v0 + v1, sq = v0 * v0 + v1 * v1;
    #pragma unroll
    for (int o = 16; o >= 1; o >>= 1) {
        s  += __shfl_xor(s, o, 32);
        sq += __shfl_xor(sq, o, 32);
    }
    __shared__ float ss[8], sv[8];
    int wid = tid >> 5, ln_ = tid & 31;
    if (ln_ == 0) { ss[wid] = s; sv[wid] = sq; }
    __syncthreads();
    if (tid == 0) {
        float a = 0.f, c = 0.f;
        #pragma unroll
        for (int i = 0; i < 8; i++) { a += ss[i]; c += sv[i]; }
        ss[0] = a; sv[0] = c;
    }
    __syncthreads();
    float mu  = ss[0] * (1.f / En);
    float var = sv[0] * (1.f / En) - mu * mu;
    float is  = rsqrtf(var + 1e-5f);
    float y0 = (v0 - mu) * is * g[tid]       + bb[tid];
    float y1 = (v1 - mu) * is * g[tid + 256] + bb[tid + 256];
    outf[(size_t)row * En + tid]       = y0;
    outf[(size_t)row * En + tid + 256] = y1;
    outb[(size_t)row * En + tid]       = f2bf(y0);
    outb[(size_t)row * En + tid + 256] = f2bf(y1);
}

// ---------------------------------------------------------------------------
// Host orchestration
// ---------------------------------------------------------------------------
extern "C" void kernel_launch(void* const* d_in, const int* in_sizes, int n_in,
                              void* d_out, int out_size, void* d_ws, size_t ws_size,
                              hipStream_t stream) {
    (void)in_sizes; (void)n_in; (void)out_size; (void)ws_size;

    const float* feat = (const float*)d_in[0];
    /* d_in[1] = positions (unused, faithful to reference) */
    const float* Wp   = (const float*)d_in[2];
    const float* bp   = (const float*)d_in[3];
    const float* pe   = (const float*)d_in[4];
    const float* Wqkv = (const float*)d_in[5];
    const float* bqkv = (const float*)d_in[6];
    const float* Wo   = (const float*)d_in[7];
    const float* bo   = (const float*)d_in[8];
    const float* ln1g = (const float*)d_in[9];
    const float* ln1b = (const float*)d_in[10];
    const float* W1   = (const float*)d_in[11];
    const float* b1   = (const float*)d_in[12];
    const float* W2   = (const float*)d_in[13];
    const float* b2   = (const float*)d_in[14];
    const float* ln2g = (const float*)d_in[15];
    const float* ln2b = (const float*)d_in[16];
    // d_in[17]/d_in[18]: past/future frames (128/128, baked into WIN)
    float* out = (float*)d_out;

    char* ws = (char*)d_ws;
    size_t off = 0;
    auto alloc = [&](size_t bytes) -> void* {
        void* p = ws + off;
        off += (bytes + 255) & ~(size_t)255;
        return p;
    };

    float*          xf    = (float*)alloc((size_t)Mn * En * 4);
    float*          tmp   = (float*)alloc((size_t)Mn * En * 4);
    unsigned short* xb    = (unsigned short*)alloc((size_t)Mn * En * 2);
    unsigned short* attnb = (unsigned short*)alloc((size_t)Mn * En * 2);
    unsigned short* qkvb  = (unsigned short*)alloc((size_t)Mn * 3 * En * 2);
    unsigned short* hb    = (unsigned short*)alloc((size_t)Mn * FFn * 2);
    unsigned short* WqkvB = (unsigned short*)alloc((size_t)Ln * 3 * En * En * 2);
    unsigned short* WoB   = (unsigned short*)alloc((size_t)Ln * En * En * 2);
    unsigned short* W1B   = (unsigned short*)alloc((size_t)Ln * FFn * En * 2);
    unsigned short* W2B   = (unsigned short*)alloc((size_t)Ln * En * FFn * 2);

    auto cvt = [&](const float* s, unsigned short* d, int n) {
        cvt_bf16_kernel<<<(n + 255) / 256, 256, 0, stream>>>(s, d, n);
    };
    cvt(Wqkv, WqkvB, Ln * 3 * En * En);
    cvt(Wo,   WoB,   Ln * En * En);
    cvt(W1,   W1B,   Ln * FFn * En);
    cvt(W2,   W2B,   Ln * En * FFn);

    proj_kernel<<<Mn, 256, 0, stream>>>(feat, Wp, bp, pe, xf, xb);

    for (int l = 0; l < Ln; l++) {
        // qkv = x @ Wqkv^T + bqkv  (bf16 out for attention)
        gemm_bf16_kernel<false, false, false, true>
            <<<dim3(3 * En / 128, Mn / 128), 256, 0, stream>>>(
                xb, WqkvB + (size_t)l * 3 * En * En, bqkv + (size_t)l * 3 * En,
                nullptr, nullptr, qkvb, Mn, 3 * En, En);

        // banded flash attention
        attn_kernel<<<dim3(Tn / 16, Hn, Bn), 32, 0, stream>>>(qkvb, attnb);

        // tmp = x + attn @ Wo^T + bo   (f32 for LN)
        gemm_bf16_kernel<false, true, true, false>
            <<<dim3(En / 128, Mn / 128), 256, 0, stream>>>(
                attnb, WoB + (size_t)l * En * En, bo + (size_t)l * En,
                xf, tmp, nullptr, Mn, En, En);

        ln_kernel<<<Mn, 256, 0, stream>>>(tmp, ln1g + (size_t)l * En,
                                          ln1b + (size_t)l * En, xf, xb);

        // h = relu(x @ W1^T + b1)  (bf16)
        gemm_bf16_kernel<true, false, false, true>
            <<<dim3(FFn / 128, Mn / 128), 256, 0, stream>>>(
                xb, W1B + (size_t)l * FFn * En, b1 + (size_t)l * FFn,
                nullptr, nullptr, hb, Mn, FFn, En);

        // tmp = x + h @ W2^T + b2  (f32)
        gemm_bf16_kernel<false, true, true, false>
            <<<dim3(En / 128, Mn / 128), 256, 0, stream>>>(
                hb, W2B + (size_t)l * En * FFn, b2 + (size_t)l * En,
                xf, tmp, nullptr, Mn, En, FFn);

        float* ldst = (l == Ln - 1) ? out : xf;
        ln_kernel<<<Mn, 256, 0, stream>>>(tmp, ln2g + (size_t)l * En,
                                          ln2b + (size_t)l * En, ldst, xb);
    }
}
